// CSA_Layer_15788299780822
// MI455X (gfx1250) — matrix operand order repaired and checked by reading, not compiled, not run on hardware
//
#include <hip/hip_runtime.h>

typedef __bf16 bf16;
typedef __attribute__((ext_vector_type(4)))  bf16  v4bf;
typedef __attribute__((ext_vector_type(8)))  bf16  v8bf;
typedef __attribute__((ext_vector_type(16))) bf16  v16bf;
typedef __attribute__((ext_vector_type(4)))  float v4f;
typedef __attribute__((ext_vector_type(8)))  float v8f;

// Problem constants (reference: B=16, C=128, N=16384)
constexpr int C      = 128;
constexpr int NPTS   = 16384;
constexpr int NTILE  = 128;              // n-columns per block
constexpr int WSTR   = 136;              // padded LDS row stride (bf16 elems): 272B = 17x16B
constexpr int WELE   = C * WSTR;         // elems per 128-row LDS matrix
constexpr size_t LDS_BYTES =
    size_t(WELE) * 2 * 4                 // wk, wv, pos, feat (bf16)
  + size_t(C) * 4                        // wq column sums (f32)
  + size_t(NTILE) * 4;                   // sq[n] (f32)

__device__ __forceinline__ v16bf ldpair(const bf16* p0, const bf16* p1) {
  v8bf lo = *(const v8bf*)p0;
  v8bf hi = *(const v8bf*)p1;
  return __builtin_shufflevector(lo, hi, 0,1,2,3,4,5,6,7,8,9,10,11,12,13,14,15);
}

__global__ __launch_bounds__(256, 1)
void csa_fused_kernel(const float* __restrict__ feature,
                      const float* __restrict__ position,
                      const float* __restrict__ Wq,
                      const float* __restrict__ Wk,
                      const float* __restrict__ Wv,
                      float* __restrict__ out) {
  extern __shared__ __align__(16) char smem_raw[];
  bf16*  s_wk  = (bf16*)smem_raw;
  bf16*  s_wv  = s_wk  + WELE;
  bf16*  s_pos = s_wv  + WELE;            // transposed: s_pos[n][c]
  bf16*  s_ft  = s_pos + WELE;            // transposed: s_ft [n][c]
  float* s_wq  = (float*)(s_ft + WELE);   // colsum(Wq)[c]
  float* s_sq  = s_wq + C;                // sq[n]

  const int t    = threadIdx.x;
  const int lane = t & 31;
  const int wave = t >> 5;
  const int b    = blockIdx.x >> 7;       // batch
  const int n0   = (blockIdx.x & 127) * NTILE;

  const float* posg = position + (size_t)b * C * NPTS + n0;
  const float* ftg  = feature  + (size_t)b * C * NPTS + n0;

  // ---- Stage Wk/Wv to LDS as bf16 (coalesced b128 reads) ----
  #pragma unroll
  for (int i = 0; i < 16; ++i) {
    int base = i * 1024 + t * 4;
    int r = base >> 7, c = base & 127;
    v4f k4 = *(const v4f*)(Wk + base);
    v4f w4 = *(const v4f*)(Wv + base);
    v4bf kb = { (bf16)k4[0], (bf16)k4[1], (bf16)k4[2], (bf16)k4[3] };
    v4bf wb = { (bf16)w4[0], (bf16)w4[1], (bf16)w4[2], (bf16)w4[3] };
    *(v4bf*)(s_wk + r * WSTR + c) = kb;
    *(v4bf*)(s_wv + r * WSTR + c) = wb;
  }
  // ---- colsum(Wq): the whole q-GEMM collapses to this vector ----
  if (t < C) {
    float a = 0.f;
    for (int o = 0; o < C; ++o) a += Wq[o * C + t];   // coalesced across threads
    s_wq[t] = a;
  }
  // ---- Stage pos/feat tile transposed [n][c] as bf16 ----
  #pragma unroll
  for (int i = 0; i < 16; ++i) {
    int base = i * 1024 + t * 4;
    int c = base >> 7, j = base & 127;
    v4f p4 = *(const v4f*)(posg + (size_t)c * NPTS + j);
    v4f f4 = *(const v4f*)(ftg  + (size_t)c * NPTS + j);
    #pragma unroll
    for (int u = 0; u < 4; ++u) {
      s_pos[(j + u) * WSTR + c] = (bf16)p4[u];
      s_ft [(j + u) * WSTR + c] = (bf16)f4[u];
    }
  }
  __syncthreads();

  // ---- sq[n] = dot(colsum(Wq), pos[:,n]) ----
  if (t < NTILE) {
    float a = 0.f;
    #pragma unroll
    for (int c8 = 0; c8 < 16; ++c8) {
      v8bf pv = *(const v8bf*)(s_pos + t * WSTR + c8 * 8);
      #pragma unroll
      for (int u = 0; u < 8; ++u) a += s_wq[c8 * 8 + u] * (float)pv[u];
    }
    s_sq[t] = a;
  }
  __syncthreads();

  // ---- WMMA: each wave computes 128 channels x 16 columns ----
  const int col = lane & 15;   // n within subtile (also A-row lane id)
  const int hi  = lane >> 4;   // K-half selector / output-row half
  const int jn  = wave * 16;   // this wave's n-subtile base

  v8f acck[8], accv[8];
  {
    const bf16* xb = s_pos + (jn + col) * WSTR + hi * 8;
    v16bf bfrag[4];
    #pragma unroll
    for (int kt = 0; kt < 4; ++kt)
      bfrag[kt] = ldpair(xb + kt * 32, xb + kt * 32 + 16);
    #pragma unroll
    for (int ot = 0; ot < 8; ++ot) {
      v8f a = {};
      const bf16* wb = s_wk + (ot * 16 + col) * WSTR + hi * 8;
      #pragma unroll
      for (int kt = 0; kt < 4; ++kt) {
        v16bf af = ldpair(wb + kt * 32, wb + kt * 32 + 16);
        a = __builtin_amdgcn_wmma_f32_16x16x32_bf16(
              false, af, false, bfrag[kt], (short)0, a, false, false);
      }
      acck[ot] = a;
    }
  }
  {
    const bf16* xb = s_ft + (jn + col) * WSTR + hi * 8;
    v16bf bfrag[4];
    #pragma unroll
    for (int kt = 0; kt < 4; ++kt)
      bfrag[kt] = ldpair(xb + kt * 32, xb + kt * 32 + 16);
    #pragma unroll
    for (int ot = 0; ot < 8; ++ot) {
      v8f a = {};
      const bf16* wb = s_wv + (ot * 16 + col) * WSTR + hi * 8;
      #pragma unroll
      for (int kt = 0; kt < 4; ++kt) {
        v16bf af = ldpair(wb + kt * 32, wb + kt * 32 + 16);
        a = __builtin_amdgcn_wmma_f32_16x16x32_bf16(
              false, af, false, bfrag[kt], (short)0, a, false, false);
      }
      accv[ot] = a;
    }
  }

  // ---- Channel-axis normalize + softmax: lane pair (l, l^16) holds all 128 channels of column n ----
  const float sqv = s_sq[jn + col];
  float S = 0.f;
  #pragma unroll
  for (int ot = 0; ot < 8; ++ot) {
    #pragma unroll
    for (int g = 0; g < 8; ++g) S += acck[ot][g];
  }
  S += __shfl_xor(S, 16, 32);
  const float inv = 1.0f / (1e-9f + sqv * S);   // energy/(1e-9 + sum energy), energy = sq*k

  float mx = -3.4e38f;
  #pragma unroll
  for (int ot = 0; ot < 8; ++ot) {
    #pragma unroll
    for (int g = 0; g < 8; ++g) {
      float e = sqv * acck[ot][g] * inv;
      acck[ot][g] = e;
      mx = fmaxf(mx, e);
    }
  }
  mx = fmaxf(mx, __shfl_xor(mx, 16, 32));

  float se = 0.f;
  #pragma unroll
  for (int ot = 0; ot < 8; ++ot) {
    #pragma unroll
    for (int g = 0; g < 8; ++g) {
      float ex = __expf(acck[ot][g] - mx);
      acck[ot][g] = ex;
      se += ex;
    }
  }
  se += __shfl_xor(se, 16, 32);
  const float rs = 1.0f / se;

  // ---- out = attn * v + feature ----
  const int nabs = n0 + jn + col;
  #pragma unroll
  for (int ot = 0; ot < 8; ++ot) {
    #pragma unroll
    for (int g = 0; g < 8; ++g) {
      int o = ot * 16 + hi * 8 + g;
      size_t idx = ((size_t)b * C + o) * NPTS + nabs;
      out[idx] = acck[ot][g] * rs * accv[ot][g] + feature[idx];
    }
  }
}

extern "C" void kernel_launch(void* const* d_in, const int* in_sizes, int n_in,
                              void* d_out, int out_size, void* d_ws, size_t ws_size,
                              hipStream_t stream) {
  const float* feature  = (const float*)d_in[0];
  const float* position = (const float*)d_in[1];
  const float* Wq       = (const float*)d_in[2];
  const float* Wk       = (const float*)d_in[3];
  const float* Wv       = (const float*)d_in[4];
  float*       out      = (float*)d_out;

  (void)in_sizes; (void)n_in; (void)out_size; (void)d_ws; (void)ws_size;

  hipFuncSetAttribute((const void*)csa_fused_kernel,
                      hipFuncAttributeMaxDynamicSharedMemorySize,
                      (int)LDS_BYTES);

  const int blocks = 16 * (NPTS / NTILE);   // B * N/NTILE = 2048
  csa_fused_kernel<<<dim3(blocks), dim3(256), LDS_BYTES, stream>>>(
      feature, position, Wq, Wk, Wv, out);
}
